// BidirectionalCrossAttentionLayer_9672266351331
// MI455X (gfx1250) — matrix operand order repaired and checked by reading, not compiled, not run on hardware
//
#include <hip/hip_runtime.h>
#include <hip/hip_bf16.h>

// ---------------------------------------------------------------------------
// Types for CDNA5 WMMA (wave32): v_wmma_f32_16x16x32_bf16
// ---------------------------------------------------------------------------
typedef __bf16 bf16;
typedef __attribute__((ext_vector_type(16))) __bf16 bf16x16;
typedef __attribute__((ext_vector_type(8)))  float  f32x8;

struct alignas(16) B16 { unsigned int a, b, c, d; };   // POD 16-byte chunk

union Frag {
    bf16x16 v;
    B16     q[2];
};

__device__ __forceinline__ f32x8 wmma_bf16(bf16x16 a, bf16x16 b, f32x8 c) {
    // (neg_a, A, neg_b, B, c_mod, C, reuse_a, reuse_b)
    return __builtin_amdgcn_wmma_f32_16x16x32_bf16(false, a, false, b,
                                                   (short)0, c, false, false);
}

// CDNA5 async global->LDS copy (ASYNCcnt-tracked), 16B per lane.
// vdst = per-lane LDS byte offset, vaddr = per-lane 64-bit global address.
__device__ __forceinline__ void async_copy_b128(unsigned ldsoff, const void* g) {
    unsigned long long ga = (unsigned long long)g;
    asm volatile("global_load_async_to_lds_b128 %0, %1, off"
                 :: "v"(ldsoff), "v"(ga) : "memory");
}
__device__ __forceinline__ void wait_async_5() {
    asm volatile("s_wait_asynccnt 0x5" ::: "memory");
}
__device__ __forceinline__ void wait_async_0() {
    asm volatile("s_wait_asynccnt 0x0" ::: "memory");
}
// Generic (flat) LDS pointer -> 32-bit LDS offset (LDS aperture keeps offset
// in addr[31:0] per CDNA5 aperture rules).
__device__ __forceinline__ unsigned lds_off(const void* p) {
    return (unsigned)(unsigned long long)p;
}

// Problem constants
#define C_DIM   1024
#define C_HID   4096
#define C_B     64
#define C_HEADS 16
#define C_DH    64
#define C_NT    256
#define C_NI    576

// ---------------------------------------------------------------------------
// Elementwise f32 -> bf16 convert
// ---------------------------------------------------------------------------
__global__ __launch_bounds__(256)
void cvt_kernel(const float* __restrict__ in, bf16* __restrict__ out, size_t n) {
    size_t i = (size_t)blockIdx.x * 256 + threadIdx.x;
    if (i < n) out[i] = (bf16)in[i];
}

// Transpose + convert: in f32 [K][N] -> out bf16 [N][K]   (grid = K*N/256 exact)
__global__ __launch_bounds__(256)
void tcvt_kernel(const float* __restrict__ in, bf16* __restrict__ out, int K, int N) {
    size_t i = (size_t)blockIdx.x * 256 + threadIdx.x;  // i = n*K + k
    size_t k = i % (size_t)K;
    size_t n = i / (size_t)K;
    out[i] = (bf16)in[k * (size_t)N + n];
}

// ---------------------------------------------------------------------------
// LayerNorm over last dim (1024), f32 in -> bf16 out. One block per row.
// ---------------------------------------------------------------------------
__global__ __launch_bounds__(256)
void ln_kernel(const float* __restrict__ x, const float* __restrict__ g,
               const float* __restrict__ bta, bf16* __restrict__ out) {
    int row = blockIdx.x;
    int tid = threadIdx.x;
    const float* xr = x + (size_t)row * C_DIM;
    float v[4];
    float s = 0.f, s2 = 0.f;
#pragma unroll
    for (int i = 0; i < 4; ++i) {
        v[i] = xr[tid + i * 256];
        s += v[i];
        s2 += v[i] * v[i];
    }
#pragma unroll
    for (int o = 16; o > 0; o >>= 1) {
        s  += __shfl_xor(s, o, 32);
        s2 += __shfl_xor(s2, o, 32);
    }
    __shared__ float rs[8], rs2[8];
    int wave = tid >> 5, lane = tid & 31;
    if (lane == 0) { rs[wave] = s; rs2[wave] = s2; }
    __syncthreads();
    s = 0.f; s2 = 0.f;
#pragma unroll
    for (int w = 0; w < 8; ++w) { s += rs[w]; s2 += rs2[w]; }
    float mean = s * (1.f / C_DIM);
    float var  = s2 * (1.f / C_DIM) - mean * mean;
    float rinv = rsqrtf(var + 1e-5f);
    bf16* orow = out + (size_t)row * C_DIM;
#pragma unroll
    for (int i = 0; i < 4; ++i) {
        int c = tid + i * 256;
        orow[c] = (bf16)((v[i] - mean) * rinv * g[c] + bta[c]);
    }
}

// ---------------------------------------------------------------------------
// GEMM: C[M,N] = A[M,K] (bf16, row-major) @ Bt[N,K] (bf16, W transposed)
// Block tile 64x256, BK=32, double-buffered LDS filled with CDNA5 async
// global->LDS b128 copies (ASYNCcnt). 8 waves of 32x64 (2x4 WMMA tiles).
// Epilogue: +bias, optional exact GELU, +residual(f32), write f32 and/or bf16.
// vtrans==1: write bf16 out in attention-V layout Vt[((b*16+h)*64+d)*NKV + n].
// ---------------------------------------------------------------------------
__global__ __launch_bounds__(256)
void gemm_kernel(const bf16* __restrict__ A, const bf16* __restrict__ Bt,
                 const float* __restrict__ bias, const float* __restrict__ resid,
                 float* __restrict__ outF, bf16* __restrict__ outB,
                 int M, int N, int K, int gelu, int vtrans, int NKV) {
    __shared__ bf16 As[2][64 * 32];    // 2 x 4 KB
    __shared__ bf16 Bs[2][256 * 32];   // 2 x 16 KB

    int tid  = threadIdx.x;
    int lane = tid & 31, wave = tid >> 5;
    int wm = wave & 1, wn = wave >> 1;           // 2 waves in M, 4 in N
    int half = lane >> 4, l16 = lane & 15;
    int mBase = blockIdx.y * 64;
    int nBase = blockIdx.x * 256;

    f32x8 acc[2][4];
#pragma unroll
    for (int i = 0; i < 2; ++i)
#pragma unroll
        for (int j = 0; j < 4; ++j) acc[i][j] = {};

    // staging geometry: A -> one b128/thread, B -> four b128/thread (row tid)
    int arow = tid >> 2, acol = (tid & 3) << 3;
    const bf16* aG = A  + (size_t)(mBase + arow) * K + acol;
    const bf16* bG = Bt + (size_t)(nBase + tid) * K;
    unsigned aL[2], bL[2];
#pragma unroll
    for (int s = 0; s < 2; ++s) {
        aL[s] = lds_off(&As[s][arow * 32 + acol]);
        bL[s] = lds_off(&Bs[s][tid * 32]);
    }

    int nstage = K >> 5;
    // prefetch stage 0 into buffer 0 (5 async ops per wave)
    async_copy_b128(aL[0], aG);
#pragma unroll
    for (int c = 0; c < 4; ++c)
        async_copy_b128(bL[0] + c * 16, bG + c * 8);

    for (int k = 0; k < nstage; ++k) {
        int cur = k & 1;
        if (k + 1 < nstage) {
            int nxt = cur ^ 1;
            const bf16* aGn = aG + (size_t)(k + 1) * 32;
            const bf16* bGn = bG + (size_t)(k + 1) * 32;
            async_copy_b128(aL[nxt], aGn);
#pragma unroll
            for (int c = 0; c < 4; ++c)
                async_copy_b128(bL[nxt] + c * 16, bGn + c * 8);
            wait_async_5();               // stage k complete (in-order retire)
        } else {
            wait_async_0();
        }
        __syncthreads();                  // stage-k data visible block-wide

        Frag a[2], b[4];
#pragma unroll
        for (int i = 0; i < 2; ++i) {
            int r = wm * 32 + i * 16 + l16;
            a[i].q[0] = *(const B16*)(&As[cur][r * 32 + half * 8]);
            a[i].q[1] = *(const B16*)(&As[cur][r * 32 + 16 + half * 8]);
        }
#pragma unroll
        for (int j = 0; j < 4; ++j) {
            int r = wn * 64 + j * 16 + l16;
            b[j].q[0] = *(const B16*)(&Bs[cur][r * 32 + half * 8]);
            b[j].q[1] = *(const B16*)(&Bs[cur][r * 32 + 16 + half * 8]);
        }
#pragma unroll
        for (int i = 0; i < 2; ++i)
#pragma unroll
            for (int j = 0; j < 4; ++j)
                acc[i][j] = wmma_bf16(a[i].v, b[j].v, acc[i][j]);
        __syncthreads();                  // safe to overwrite buf `cur` later
    }

    // epilogue: C layout: VGPR r -> M = (lane/16)*8 + r, N = lane%16
#pragma unroll
    for (int i = 0; i < 2; ++i) {
#pragma unroll
        for (int j = 0; j < 4; ++j) {
            int gm0 = mBase + wm * 32 + i * 16 + half * 8;
            int gn  = nBase + wn * 64 + j * 16 + l16;
            float bval = bias ? bias[gn] : 0.f;
#pragma unroll
            for (int r = 0; r < 8; ++r) {
                int gm = gm0 + r;
                float v = acc[i][j][r] + bval;
                if (gelu) v = 0.5f * v * (1.f + erff(v * 0.7071067811865475f));
                if (resid) v += resid[(size_t)gm * N + gn];
                if (outF) outF[(size_t)gm * N + gn] = v;
                if (outB) {
                    if (vtrans) {
                        int n = gm >> 6, bb = gm & 63;   // row = n*B + b, B=64
                        int hh = gn >> 6, dd = gn & 63;  // col = h*64 + d
                        outB[((size_t)((bb * C_HEADS + hh) * C_DH + dd)) * NKV + n] = (bf16)v;
                    } else {
                        outB[(size_t)gm * N + gn] = (bf16)v;
                    }
                }
            }
        }
    }
}

// ---------------------------------------------------------------------------
// Fused attention per (b, h, 32-query tile).
//   Q  : bf16 [n, b, dim]   (query projection)
//   Kp : bf16 [m, b, dim]   (key projection; [m][dh] slice is WMMA-B friendly)
//   Vt : bf16 [((b*16+h)*64+d)*Mk + m]  (V projection pre-transposed)
//   AO : bf16 [n, b, dim]
// 8 waves: 2 query subtiles x 4 key strips (S phase) / 4 d-tiles (O phase).
// Dynamic LDS: Qs (32x64 bf16) + Ss (32 x SW f32).
// ---------------------------------------------------------------------------
__global__ __launch_bounds__(256)
void attn_kernel(const bf16* __restrict__ Q, const bf16* __restrict__ Kp,
                 const bf16* __restrict__ Vt, bf16* __restrict__ AO,
                 int Nq, int Mk, int SW, float scale) {
    extern __shared__ char smem[];
    bf16*  Qs = (bf16*)smem;                 // 32*64*2 = 4096 B
    float* Ss = (float*)(smem + 4096);       // 32*SW*4 B

    int tid  = threadIdx.x;
    int lane = tid & 31, wave = tid >> 5;
    int half = lane >> 4, l16 = lane & 15;
    int bh = blockIdx.y;
    int b  = bh >> 4, h = bh & 15;
    int q0 = blockIdx.x * 32;

    // stage Q tile [32 x 64]
    {
        int row = tid >> 3, seg = (tid & 7) << 3;
        *(B16*)(&Qs[row * 64 + seg]) =
            *(const B16*)(&Q[(((size_t)(q0 + row)) * C_B + b) * C_DIM + h * C_DH + seg]);
    }
    __syncthreads();

    int qs    = wave & 1;        // query subtile (16 rows)
    int strip = wave >> 1;       // key strip (Mk/4 keys)
    int tilesPerStrip = Mk >> 6; // (Mk/4)/16

    // A fragments of Q (dh=64 -> two K=32 chunks), reused for all key tiles
    Frag aq[2];
#pragma unroll
    for (int c = 0; c < 2; ++c) {
        int r = qs * 16 + l16;
        aq[c].q[0] = *(const B16*)(&Qs[r * 64 + c * 32 + half * 8]);
        aq[c].q[1] = *(const B16*)(&Qs[r * 64 + c * 32 + 16 + half * 8]);
    }

    // S = Q @ K^T (scaled), scattered into LDS
    for (int it = 0; it < tilesPerStrip; ++it) {
        int mt = (strip * tilesPerStrip + it) * 16;
        int mrow = mt + l16;
        const bf16* kbase = Kp + ((size_t)mrow * C_B + b) * C_DIM + h * C_DH;
        Frag bk[2];
#pragma unroll
        for (int c = 0; c < 2; ++c) {
            bk[c].q[0] = *(const B16*)(kbase + c * 32 + half * 8);
            bk[c].q[1] = *(const B16*)(kbase + c * 32 + 16 + half * 8);
        }
        f32x8 s = {};
        s = wmma_bf16(aq[0].v, bk[0].v, s);
        s = wmma_bf16(aq[1].v, bk[1].v, s);
        int srow0 = qs * 16 + half * 8;
#pragma unroll
        for (int r = 0; r < 8; ++r)
            Ss[(size_t)(srow0 + r) * SW + mt + l16] = s[r] * scale;
    }
    __syncthreads();

    // softmax: 8 threads per query row
    {
        int row = tid >> 3, sub = tid & 7;
        float mx = -3.4e38f;
        for (int m = sub; m < Mk; m += 8) mx = fmaxf(mx, Ss[(size_t)row * SW + m]);
#pragma unroll
        for (int o = 1; o < 8; o <<= 1) mx = fmaxf(mx, __shfl_xor(mx, o, 8));
        float sum = 0.f;
        for (int m = sub; m < Mk; m += 8) {
            float e = __expf(Ss[(size_t)row * SW + m] - mx);
            Ss[(size_t)row * SW + m] = e;
            sum += e;
        }
#pragma unroll
        for (int o = 1; o < 8; o <<= 1) sum += __shfl_xor(sum, o, 8);
        float inv = 1.f / sum;
        for (int m = sub; m < Mk; m += 8) Ss[(size_t)row * SW + m] *= inv;
    }
    __syncthreads();

    // O = P @ V   (wave: qs x d-tile), K dimension = Mk in chunks of 32
    int dt = wave >> 1;                  // 0..3 -> d subtile of 16
    int prow = qs * 16 + l16;
    const bf16* vbase = Vt + ((size_t)((b * C_HEADS + h) * C_DH + dt * 16 + l16)) * Mk;
    f32x8 o = {};
    for (int kc = 0; kc < Mk; kc += 32) {
        Frag pa, vb;
#pragma unroll
        for (int t = 0; t < 8; ++t) {
            pa.v[t]     = (bf16)Ss[(size_t)prow * SW + kc + half * 8 + t];
            pa.v[8 + t] = (bf16)Ss[(size_t)prow * SW + kc + 16 + half * 8 + t];
        }
        vb.q[0] = *(const B16*)(vbase + kc + half * 8);
        vb.q[1] = *(const B16*)(vbase + kc + 16 + half * 8);
        o = wmma_bf16(pa.v, vb.v, o);
    }
    int gq0 = q0 + qs * 16 + half * 8;
    int dd  = dt * 16 + l16;
#pragma unroll
    for (int r = 0; r < 8; ++r)
        AO[(((size_t)(gq0 + r)) * C_B + b) * C_DIM + h * C_DH + dd] = (bf16)o[r];
}

// ---------------------------------------------------------------------------
// Host-side orchestration
// ---------------------------------------------------------------------------
static void run_phase(hipStream_t stream, char* ws,
                      const float* qin, const float* kvin,
                      int MQ, int MKV, int Nq, int Mk,
                      const float* lng, const float* lnb,
                      const float* wq, const float* wk, const float* wv,
                      const float* wo, const float* bo,
                      const float* fng, const float* fnb,
                      const float* w1, const float* b1,
                      const float* w2, const float* b2,
                      float* outp) {
    const size_t SB = (size_t)36864 * 1024 * 2;   // max bf16 [rows,1024] buffer
    bf16* Qb  = (bf16*)(ws + 0 * SB);
    bf16* Kb  = (bf16*)(ws + 1 * SB);
    bf16* Vb  = (bf16*)(ws + 2 * SB);
    bf16* AOb = (bf16*)(ws + 3 * SB);
    bf16* Hb  = (bf16*)(ws + 0 * SB);             // reuses Q..AO region (FFN hidden)
    bf16* LNb = (bf16*)(ws + 4 * SB);
    bf16* KVb = (bf16*)(ws + 5 * SB);
    float* RES = (float*)(ws + 6 * SB);
    char* wbase = ws + 6 * SB + (size_t)36864 * 1024 * 4;
    bf16* WQ  = (bf16*)wbase;
    bf16* WK  = WQ + (size_t)1024 * 1024;
    bf16* WV  = WK + (size_t)1024 * 1024;
    bf16* WO  = WV + (size_t)1024 * 1024;
    bf16* W1T = WO + (size_t)1024 * 1024;
    bf16* W2T = W1T + (size_t)1024 * 4096;

    // convert weights to transposed bf16
    tcvt_kernel<<<(1024 * 1024) / 256, 256, 0, stream>>>(wq, WQ, 1024, 1024);
    tcvt_kernel<<<(1024 * 1024) / 256, 256, 0, stream>>>(wk, WK, 1024, 1024);
    tcvt_kernel<<<(1024 * 1024) / 256, 256, 0, stream>>>(wv, WV, 1024, 1024);
    tcvt_kernel<<<(1024 * 1024) / 256, 256, 0, stream>>>(wo, WO, 1024, 1024);
    tcvt_kernel<<<(1024 * 4096) / 256, 256, 0, stream>>>(w1, W1T, 1024, 4096);
    tcvt_kernel<<<(1024 * 4096) / 256, 256, 0, stream>>>(w2, W2T, 4096, 1024);

    // LN(query input) -> bf16 ; convert kv input -> bf16
    ln_kernel<<<MQ, 256, 0, stream>>>(qin, lng, lnb, LNb);
    size_t nkv = (size_t)MKV * 1024;
    cvt_kernel<<<(unsigned)(nkv / 256), 256, 0, stream>>>(kvin, KVb, nkv);

    // projections (block tile 64 x 256)
    gemm_kernel<<<dim3(1024 / 256, MQ / 64), 256, 0, stream>>>(
        LNb, WQ, nullptr, nullptr, nullptr, Qb, MQ, 1024, 1024, 0, 0, 0);
    gemm_kernel<<<dim3(1024 / 256, MKV / 64), 256, 0, stream>>>(
        KVb, WK, nullptr, nullptr, nullptr, Kb, MKV, 1024, 1024, 0, 0, 0);
    gemm_kernel<<<dim3(1024 / 256, MKV / 64), 256, 0, stream>>>(
        KVb, WV, nullptr, nullptr, nullptr, Vb, MKV, 1024, 1024, 0, 1, Mk);

    // attention
    int SW = Mk + 8;
    size_t shmem = 4096 + (size_t)32 * SW * 4;
    attn_kernel<<<dim3(Nq / 32, C_B * C_HEADS), 256, shmem, stream>>>(
        Qb, Kb, Vb, AOb, Nq, Mk, SW, 0.125f);

    // output projection + residual  -> RES (f32)
    gemm_kernel<<<dim3(1024 / 256, MQ / 64), 256, 0, stream>>>(
        AOb, WO, bo, qin, RES, nullptr, MQ, 1024, 1024, 0, 0, 0);

    // FFN: LN -> GELU(W1) -> W2 + residual -> out
    ln_kernel<<<MQ, 256, 0, stream>>>(RES, fng, fnb, LNb);
    gemm_kernel<<<dim3(4096 / 256, MQ / 64), 256, 0, stream>>>(
        LNb, W1T, b1, nullptr, nullptr, Hb, MQ, 4096, 1024, 1, 0, 0);
    gemm_kernel<<<dim3(1024 / 256, MQ / 64), 256, 0, stream>>>(
        Hb, W2T, b2, RES, outp, nullptr, MQ, 1024, 4096, 0, 0, 0);
}

extern "C" void kernel_launch(void* const* d_in, const int* in_sizes, int n_in,
                              void* d_out, int out_size, void* d_ws, size_t ws_size,
                              hipStream_t stream) {
    const float* tokens = (const float*)d_in[0];
    const float* image  = (const float*)d_in[1];
    const float* tn_g = (const float*)d_in[2];
    const float* tn_b = (const float*)d_in[3];
    const float* t2i_wq = (const float*)d_in[4];
    const float* t2i_wk = (const float*)d_in[5];
    const float* t2i_wv = (const float*)d_in[6];
    const float* t2i_wo = (const float*)d_in[7];
    const float* t2i_bo = (const float*)d_in[8];
    const float* in_g = (const float*)d_in[9];
    const float* in_b = (const float*)d_in[10];
    const float* i2t_wq = (const float*)d_in[11];
    const float* i2t_wk = (const float*)d_in[12];
    const float* i2t_wv = (const float*)d_in[13];
    const float* i2t_wo = (const float*)d_in[14];
    const float* i2t_bo = (const float*)d_in[15];
    const float* tfn_g = (const float*)d_in[16];
    const float* tfn_b = (const float*)d_in[17];
    const float* tf_w1 = (const float*)d_in[18];
    const float* tf_b1 = (const float*)d_in[19];
    const float* tf_w2 = (const float*)d_in[20];
    const float* tf_b2 = (const float*)d_in[21];
    const float* ifn_g = (const float*)d_in[22];
    const float* ifn_b = (const float*)d_in[23];
    const float* if_w1 = (const float*)d_in[24];
    const float* if_b1 = (const float*)d_in[25];
    const float* if_w2 = (const float*)d_in[26];
    const float* if_b2 = (const float*)d_in[27];

    char* ws = (char*)d_ws;
    float* out = (float*)d_out;

    (void)hipFuncSetAttribute(reinterpret_cast<const void*>(attn_kernel),
                              hipFuncAttributeMaxDynamicSharedMemorySize,
                              100 * 1024);

    // tokens branch: Q from LN(tokens), K/V from raw image_features
    run_phase(stream, ws, tokens, image,
              C_NT * C_B, C_NI * C_B, C_NT, C_NI,
              tn_g, tn_b, t2i_wq, t2i_wk, t2i_wv, t2i_wo, t2i_bo,
              tfn_g, tfn_b, tf_w1, tf_b1, tf_w2, tf_b2,
              out);

    // image branch: Q from LN(image), K/V from raw tokens
    run_phase(stream, ws, image, tokens,
              C_NI * C_B, C_NT * C_B, C_NI, C_NT,
              in_g, in_b, i2t_wq, i2t_wk, i2t_wv, i2t_wo, i2t_bo,
              ifn_g, ifn_b, if_w1, if_b1, if_w2, if_b2,
              out + (size_t)C_NT * C_B * C_DIM);
}
